// LeeNQG_21921513079479
// MI455X (gfx1250) — compile-verified, matrix operand order checked by reading, not run on hardware
//
#include <hip/hip_runtime.h>

// ---------------- model constants ----------------
#define Vn     50000
#define EMBn   300
#define Hn_    512
#define H2n_   1024
#define EDGEn  4
#define EEMBn  32
#define Bn_    16
#define TSn_   400
#define TQn    33
#define Tn_    32          // TQ-1 decode steps
#define MAXOOV 100
#define EXTn_  (Vn + MAXOOV)
#define NEG_INF (-1.0e12f)

typedef __attribute__((ext_vector_type(16))) __bf16 v16bf;
typedef __attribute__((ext_vector_type(8)))  __bf16 v8bf;
typedef __attribute__((ext_vector_type(8)))  float  v8f;

__device__ __forceinline__ __bf16 f2bf(float f) {
  unsigned u = __float_as_uint(f);
  unsigned short h = (unsigned short)(u >> 16);
  return __builtin_bit_cast(__bf16, h);
}
__device__ __forceinline__ float sigmoidf_(float x) { return 1.0f / (1.0f + __expf(-x)); }
// every lane of a wave shares per-wave indices (tid>>5); make that explicit
// so tile dispatch is SGPR-based and EXEC stays all-ones across WMMA.
__device__ __forceinline__ int uniform_(int x) { return __builtin_amdgcn_readfirstlane(x); }

// ------------------------------------------------------------------
// fragment loaders. CDNA5 16-bit A-fragment layout (CDNA5 ISA 7.12.2):
// lane (half=lane>>4, r=lane&15) holds row r, K = k0 + half*8 + {0..7, 16..23}.
// f32 source: 4x b128 + repack.  bf16 source: 2x b128, no repack.
// ------------------------------------------------------------------
__device__ __forceinline__ void load_frag_fast(v16bf& f, const float* row, int k0, int half) {
  const float* p = row + k0 + half * 8;
  float4 x0 = *(const float4*)(p);
  float4 x1 = *(const float4*)(p + 4);
  float4 x2 = *(const float4*)(p + 16);
  float4 x3 = *(const float4*)(p + 20);
  f[0]  = f2bf(x0.x); f[1]  = f2bf(x0.y); f[2]  = f2bf(x0.z); f[3]  = f2bf(x0.w);
  f[4]  = f2bf(x1.x); f[5]  = f2bf(x1.y); f[6]  = f2bf(x1.z); f[7]  = f2bf(x1.w);
  f[8]  = f2bf(x2.x); f[9]  = f2bf(x2.y); f[10] = f2bf(x2.z); f[11] = f2bf(x2.w);
  f[12] = f2bf(x3.x); f[13] = f2bf(x3.y); f[14] = f2bf(x3.z); f[15] = f2bf(x3.w);
}

__device__ __forceinline__ void load_frag_guard(v16bf& f, const float* row, int k0, int half, int K) {
  __bf16 z = f2bf(0.0f);
#pragma unroll
  for (int i = 0; i < 16; ++i) {
    int k = k0 + half * 8 + ((i < 8) ? i : (i + 8));
    f[i] = (k < K) ? f2bf(row[k]) : z;
  }
}

__device__ __forceinline__ void load_frag_bf16(v16bf& f, const __bf16* row, int k0, int half) {
  const __bf16* p = row + k0 + half * 8;
  v8bf lo = *(const v8bf*)(p);
  v8bf hi = *(const v8bf*)(p + 16);
  f = __builtin_shufflevector(lo, hi, 0, 1, 2, 3, 4, 5, 6, 7, 8, 9, 10, 11, 12, 13, 14, 15);
}

__device__ __forceinline__ void load_frag_bf16_guard(v16bf& f, const __bf16* row, int k0, int half, int K) {
  __bf16 z = f2bf(0.0f);
#pragma unroll
  for (int i = 0; i < 16; ++i) {
    int k = k0 + half * 8 + ((i < 8) ? i : (i + 8));
    f[i] = (k < K) ? row[k] : z;
  }
}

// ------------------------------------------------------------------
// branch-free k-loops, specialized on tile count JT (1..4). The guard
// dispatch happens ONCE (scalar switch), so the hot loop is pure
// loads + WMMA with pinned accumulators.
// ------------------------------------------------------------------
template <int JT>
__device__ __forceinline__ void wmma_kloop_bf(v8f acc[4], const float* arow, int half,
                                              const __bf16* b0, const __bf16* b1,
                                              const __bf16* b2, const __bf16* b3, int K) {
  const __bf16* brow[4] = {b0, b1, b2, b3};
  int KF = K & ~31;
  for (int k0 = 0; k0 < KF; k0 += 32) {
    v16bf a; load_frag_fast(a, arow, k0, half);
#pragma unroll
    for (int j = 0; j < JT; ++j) {
      v16bf b; load_frag_bf16(b, brow[j], k0, half);
      acc[j] = __builtin_amdgcn_wmma_f32_16x16x32_bf16(false, a, false, b, (short)0,
                                                       acc[j], false, false);
    }
  }
  if (KF < K) {
    v16bf a; load_frag_guard(a, arow, KF, half, K);
#pragma unroll
    for (int j = 0; j < JT; ++j) {
      v16bf b; load_frag_bf16_guard(b, brow[j], KF, half, K);
      acc[j] = __builtin_amdgcn_wmma_f32_16x16x32_bf16(false, a, false, b, (short)0,
                                                       acc[j], false, false);
    }
  }
}

template <int JT, int TRANS>
__device__ __forceinline__ void wmma_kloop_f32(v8f acc[4], const float* arow, int half,
                                               const float* b0, const float* b1,
                                               const float* b2, const float* b3,
                                               long ldb, int K) {
  const float* brow[4] = {b0, b1, b2, b3};
  int KF = K & ~31;
  for (int k0 = 0; k0 < KF; k0 += 32) {
    v16bf a; load_frag_fast(a, arow, k0, half);
#pragma unroll
    for (int j = 0; j < JT; ++j) {
      v16bf b;
      if (TRANS) load_frag_fast(b, brow[j], k0, half);
      else {
#pragma unroll
        for (int i = 0; i < 16; ++i) {
          int k = k0 + half * 8 + ((i < 8) ? i : (i + 8));
          b[i] = f2bf(brow[j][(long)k * ldb]);
        }
      }
      acc[j] = __builtin_amdgcn_wmma_f32_16x16x32_bf16(false, a, false, b, (short)0,
                                                       acc[j], false, false);
    }
  }
  if (KF < K) {
    v16bf a; load_frag_guard(a, arow, KF, half, K);
#pragma unroll
    for (int j = 0; j < JT; ++j) {
      v16bf b;
      if (TRANS) load_frag_guard(b, brow[j], KF, half, K);
      else {
        __bf16 z = f2bf(0.0f);
#pragma unroll
        for (int i = 0; i < 16; ++i) {
          int k = KF + half * 8 + ((i < 8) ? i : (i + 8));
          b[i] = (k < K) ? f2bf(brow[j][(long)k * ldb]) : z;
        }
      }
      acc[j] = __builtin_amdgcn_wmma_f32_16x16x32_bf16(false, a, false, b, (short)0,
                                                       acc[j], false, false);
    }
  }
}

// ------------------------------------------------------------------
// group-of-4 16x16 tiles: acc[j] += A(16xK,f32) * B tile (nt0+j).
// nt0 must be wave-uniform (scalarized by callers).
// ------------------------------------------------------------------
// packed-bf16-B (weight layout [N,K])
__device__ __forceinline__ void wmma_group4_bf(v8f acc[4], const float* A, long lda, int rowsA,
                                               const __bf16* Bm, long ldb,
                                               int Ntotal, int nt0, int K) {
  int lane = threadIdx.x & 31, half = lane >> 4, r = lane & 15;
  int ra = (r < rowsA) ? r : (rowsA - 1);
  const float* arow = A + (long)ra * lda;
  int tilesN = (Ntotal + 15) >> 4;
  int jmax = tilesN - nt0; if (jmax > 4) jmax = 4;
  const __bf16* brow[4];
#pragma unroll
  for (int j = 0; j < 4; ++j) {
    int nt = nt0 + ((j < jmax) ? j : 0);
    int cb = Ntotal - nt * 16;
    int rb = (r < cb) ? r : (cb - 1);
    brow[j] = Bm + ((long)nt * 16 + rb) * ldb;
  }
  switch (jmax) {
    case 4:  wmma_kloop_bf<4>(acc, arow, half, brow[0], brow[1], brow[2], brow[3], K); break;
    case 3:  wmma_kloop_bf<3>(acc, arow, half, brow[0], brow[1], brow[2], brow[3], K); break;
    case 2:  wmma_kloop_bf<2>(acc, arow, half, brow[0], brow[1], brow[2], brow[3], K); break;
    default: wmma_kloop_bf<1>(acc, arow, half, brow[0], brow[1], brow[2], brow[3], K); break;
  }
}

// f32-B: transB=1 -> weight layout [N,K]; transB=0 -> [K,N] strided gather
__device__ __forceinline__ void wmma_group4(v8f acc[4], const float* A, long lda, int rowsA,
                                            const float* Bm, long ldb, int transB,
                                            int Ntotal, int nt0, int K) {
  int lane = threadIdx.x & 31, half = lane >> 4, r = lane & 15;
  int ra = (r < rowsA) ? r : (rowsA - 1);
  const float* arow = A + (long)ra * lda;
  int tilesN = (Ntotal + 15) >> 4;
  int jmax = tilesN - nt0; if (jmax > 4) jmax = 4;
  const float* brow[4];
#pragma unroll
  for (int j = 0; j < 4; ++j) {
    int nt = nt0 + ((j < jmax) ? j : 0);
    int cb = Ntotal - nt * 16;
    int rb = (r < cb) ? r : (cb - 1);
    brow[j] = transB ? (Bm + ((long)nt * 16 + rb) * ldb) : (Bm + nt * 16 + rb);
  }
  if (transB) {
    switch (jmax) {
      case 4:  wmma_kloop_f32<4, 1>(acc, arow, half, brow[0], brow[1], brow[2], brow[3], ldb, K); break;
      case 3:  wmma_kloop_f32<3, 1>(acc, arow, half, brow[0], brow[1], brow[2], brow[3], ldb, K); break;
      case 2:  wmma_kloop_f32<2, 1>(acc, arow, half, brow[0], brow[1], brow[2], brow[3], ldb, K); break;
      default: wmma_kloop_f32<1, 1>(acc, arow, half, brow[0], brow[1], brow[2], brow[3], ldb, K); break;
    }
  } else {
    switch (jmax) {
      case 4:  wmma_kloop_f32<4, 0>(acc, arow, half, brow[0], brow[1], brow[2], brow[3], ldb, K); break;
      case 3:  wmma_kloop_f32<3, 0>(acc, arow, half, brow[0], brow[1], brow[2], brow[3], ldb, K); break;
      case 2:  wmma_kloop_f32<2, 0>(acc, arow, half, brow[0], brow[1], brow[2], brow[3], ldb, K); break;
      default: wmma_kloop_f32<1, 0>(acc, arow, half, brow[0], brow[1], brow[2], brow[3], ldb, K); break;
    }
  }
}

// C/D layout: VGPR v, lane l -> row m = v + 8*(l>=16), col n = l&15
__device__ __forceinline__ void store_tile(v8f acc, float* C, long ldc, int rows, int cols,
                                           const float* bias, int act, int addC) {
  int lane = threadIdx.x & 31;
  int half = lane >> 4;
  int n    = lane & 15;
  if (n >= cols) return;
  float bv = bias ? bias[n] : 0.0f;
#pragma unroll
  for (int v = 0; v < 8; ++v) {
    int m = v + half * 8;
    if (m < rows) {
      float x = acc[v] + bv;
      long idx = (long)m * ldc + n;
      if (addC) x += C[idx];
      if (act == 1) x = tanhf(x);
      else if (act == 2) x = sigmoidf_(x);
      C[idx] = x;
    }
  }
}

// ------------------------------------------------------------------
// generic (optionally batched) GEMM:  C = act( A @ op(B) [+ bias] [+ C_old] )
// one 16x64 output strip (4 tiles) per wave, 8 waves/block.
// BF=1: B is packed bf16 [N,K]; BF=0: B is f32 (transB selects layout).
// ------------------------------------------------------------------
template <int BF>
__global__ __launch_bounds__(256) void k_gemm_t(
    const float* A, long lda, long strideA,
    const void* Bmv, long ldb, long strideB, int transB,
    const float* bias, float* C, long ldc, long strideC,
    int M, int N, int K, int act, int addC) {
  A += (long)blockIdx.z * strideA;
  C += (long)blockIdx.z * strideC;
  int tn = (N + 15) >> 4, tm = (M + 15) >> 4;
  int gn = (tn + 3) >> 2;
  int group = uniform_((int)(blockIdx.x * (blockDim.x >> 5) + (threadIdx.x >> 5)));
  if (group >= tm * gn) return;
  int mt = group / gn, nt0 = (group % gn) * 4;
  v8f zero = {};
  v8f acc[4] = {zero, zero, zero, zero};
  if (BF) {
    const __bf16* Bm = (const __bf16*)Bmv + (long)blockIdx.z * strideB;
    wmma_group4_bf(acc, A + (long)mt * 16 * lda, lda, M - mt * 16, Bm, ldb, N, nt0, K);
  } else {
    const float* Bm = (const float*)Bmv + (long)blockIdx.z * strideB;
    wmma_group4(acc, A + (long)mt * 16 * lda, lda, M - mt * 16, Bm, ldb, transB, N, nt0, K);
  }
  int rows = M - mt * 16; if (rows > 16) rows = 16;
#pragma unroll
  for (int j = 0; j < 4; ++j) {
    int nt = nt0 + j;
    if (nt < tn) {
      int cols = N - nt * 16; if (cols > 16) cols = 16;
      store_tile(acc[j], C + (long)mt * 16 * ldc + nt * 16, ldc, rows, cols,
                 bias ? (bias + nt * 16) : nullptr, act, addC);
    }
  }
}

// ---------------- f32 -> packed bf16 weight conversion ----------------
__global__ void k_cvt_bf16(const float* in, __bf16* out, long n) {
  long i = ((long)blockIdx.x * blockDim.x + threadIdx.x) * 4;
  if (i + 3 < n) {
    float4 v = *(const float4*)(in + i);
    unsigned bx = __float_as_uint(v.x), by = __float_as_uint(v.y);
    unsigned bz = __float_as_uint(v.z), bw = __float_as_uint(v.w);
    uint2 u;
    u.x = (by & 0xFFFF0000u) | (bx >> 16);
    u.y = (bw & 0xFFFF0000u) | (bz >> 16);
    *(uint2*)(out + i) = u;
  } else {
    for (long k = i; k < n; ++k) out[k] = f2bf(in[k]);
  }
}

// ---------------- embedding / layout kernels ----------------
__global__ void k_embed_enc(const float* enc_emb, const float* ans_emb,
                            const int* input_s, const int* input_a, float* X0) {
  long i = (long)blockIdx.x * blockDim.x + threadIdx.x;
  const long tot = (long)TSn_ * Bn_ * (EMBn + EEMBn);
  if (i >= tot) return;
  int  col = (int)(i % (EMBn + EEMBn));
  long row = i / (EMBn + EEMBn);
  int  t = (int)(row / Bn_), b = (int)(row % Bn_);
  float v;
  if (col < EMBn) v = enc_emb[(long)input_s[b * TSn_ + t] * EMBn + col];
  else            v = ans_emb[(long)input_a[b * TSn_ + t] * EEMBn + (col - EMBn)];
  X0[i] = v;
}

__global__ void k_embed_dec(const float* dec_emb, const int* input_q, float* Xq) {
  long i = (long)blockIdx.x * blockDim.x + threadIdx.x;
  const long tot = (long)Tn_ * Bn_ * EMBn;
  if (i >= tot) return;
  int  col = (int)(i % EMBn);
  long r = i / EMBn;
  int  t = (int)(r / Bn_), b = (int)(r % Bn_);
  Xq[i] = dec_emb[(long)input_q[b * TQn + t] * EMBn + col];
}

__global__ void k_transpose(const float* yin, float* outbts) {  // [TS,B,D]->[B,TS,D]
  long i = (long)blockIdx.x * blockDim.x + threadIdx.x;
  const long tot = (long)TSn_ * Bn_ * H2n_;
  if (i >= tot) return;
  int  d = (int)(i % H2n_);
  long r = i / H2n_;
  int  t = (int)(r / Bn_), b = (int)(r % Bn_);
  outbts[((long)b * TSn_ + t) * H2n_ + d] = yin[i];
}

// ------------------------------------------------------------------
// persistent bi-GRU layer: 2 blocks (fwd/bwd). Hidden state in global
// workspace (all WMMA operands addrspace(1) -> global_load_b128).
// Per step: gh = h @ Wh.T + bh via bf16-weight WMMA, then fused gates.
// ------------------------------------------------------------------
__global__ __launch_bounds__(512) void k_gru_layer(
    const float* gx_f, const float* gx_b,
    const __bf16* Wh, long WhDirStride,
    const float* bh, long bhDirStride,
    float* Y, float* hfinal, float* ghh_ws, float* h_ws) {
  const int Bn = Bn_, H = Hn_, TS = TSn_;
  const int G = 3 * H;
  int dir = blockIdx.x;
  const float*  gx  = dir ? gx_b : gx_f;
  const __bf16* W   = Wh + (long)dir * WhDirStride;
  const float*  bhd = bh + (long)dir * bhDirStride;
  float* ghh = ghh_ws + (long)dir * Bn * G;
  float* h   = h_ws + (long)dir * Bn * H;
  for (int i = threadIdx.x; i < Bn * H; i += blockDim.x) h[i] = 0.0f;
  __syncthreads();
  int nwaves = blockDim.x >> 5, wave = uniform_(threadIdx.x >> 5);
  const int gn = (G / 16 + 3) >> 2;                    // 24 groups of 4 tiles
  v8f zero = {};
  for (int step = 0; step < TS; ++step) {
    int t = dir ? (TS - 1 - step) : step;
    for (int g = wave; g < gn; g += nwaves) {
      v8f acc[4] = {zero, zero, zero, zero};
      wmma_group4_bf(acc, h, H, Bn, W, H, G, g * 4, H);
#pragma unroll
      for (int j = 0; j < 4; ++j) {
        int nt = g * 4 + j;
        store_tile(acc[j], ghh + nt * 16, G, Bn, 16, bhd + nt * 16, 0, 0);
      }
    }
    __syncthreads();
    const float* gxt = gx + (long)t * Bn * G;
    float* yt = Y + (long)t * Bn * 2 * H + (long)dir * H;
    for (int i = threadIdx.x; i < Bn * H; i += blockDim.x) {
      int b = i / H, j = i % H;
      const float* ga = gxt + (long)b * G;
      const float* gb = ghh + (long)b * G;
      float r = sigmoidf_(ga[j] + gb[j]);
      float z = sigmoidf_(ga[H + j] + gb[H + j]);
      float n = tanhf(ga[2 * H + j] + r * gb[2 * H + j]);
      float hv = (1.0f - z) * n + z * h[i];
      h[i] = hv;
      yt[(long)b * 2 * H + j] = hv;
    }
    __syncthreads();
  }
  for (int i = threadIdx.x; i < Bn * H; i += blockDim.x) {
    int b = i / H, j = i % H;
    hfinal[(long)b * 2 * H + (long)dir * H + j] = h[i];
  }
}

// ---------------- softmax kernels (wave32-per-row) ----------------
__global__ void k_softmax_rows(float* X, int rows, int cols) {
  int row  = blockIdx.x * (blockDim.x >> 5) + (threadIdx.x >> 5);
  int lane = threadIdx.x & 31;
  if (row >= rows) return;
  float* p = X + (long)row * cols;
  float mx = -1e30f;
  for (int c = lane; c < cols; c += 32) mx = fmaxf(mx, p[c]);
  for (int o = 16; o; o >>= 1) mx = fmaxf(mx, __shfl_xor(mx, o, 32));
  float s = 0.0f;
  for (int c = lane; c < cols; c += 32) { float e = __expf(p[c] - mx); p[c] = e; s += e; }
  for (int o = 16; o; o >>= 1) s += __shfl_xor(s, o, 32);
  float inv = 1.0f / s;
  for (int c = lane; c < cols; c += 32) p[c] *= inv;
}

__global__ void k_softmax_dim1(float* X, int S, int D) {  // softmax over s of [B,S,D]
  int col  = blockIdx.x * (blockDim.x >> 5) + (threadIdx.x >> 5);
  int lane = threadIdx.x & 31;
  if (col >= Bn_ * D) return;
  int b = col / D, d = col % D;
  float* base = X + (long)b * S * D + d;
  float mx = -1e30f;
  for (int s = lane; s < S; s += 32) mx = fmaxf(mx, base[(long)s * D]);
  for (int o = 16; o; o >>= 1) mx = fmaxf(mx, __shfl_xor(mx, o, 32));
  float sum = 0.0f;
  for (int s = lane; s < S; s += 32) {
    float e = __expf(base[(long)s * D] - mx); base[(long)s * D] = e; sum += e;
  }
  for (int o = 16; o; o >>= 1) sum += __shfl_xor(sum, o, 32);
  float inv = 1.0f / sum;
  for (int s = lane; s < S; s += 32) base[(long)s * D] *= inv;
}

// ---------------- elementwise fusion kernels ----------------
__global__ void k_selfout(float* obts, const float* f_t, const float* g_t) {
  long i = (long)blockIdx.x * blockDim.x + threadIdx.x;
  const long tot = (long)TSn_ * Bn_ * H2n_;
  if (i >= tot) return;
  float g = g_t[i];
  obts[i] = g * f_t[i] + (1.0f - g) * obts[i];
}

__global__ void k_eraw(const float* mem, const int* input_a, float* eraw) {
  long i = (long)blockIdx.x * blockDim.x + threadIdx.x;
  const long tot = (long)TSn_ * Bn_ * H2n_;
  if (i >= tot) return;
  long r = i / H2n_;                 // r = b*TS + s
  float m = mem[i];
  eraw[i] = (input_a[r] == 0) ? m : m * m;
}

__global__ void k_encout(float* obts, const float* esoft, const float* mem) {
  long i = (long)blockIdx.x * blockDim.x + threadIdx.x;
  const long tot = (long)TSn_ * Bn_ * H2n_;
  if (i >= tot) return;
  obts[i] += esoft[i] * mem[i];
}

// ------------------------------------------------------------------
// decoder scan helper: out[16,Ntot] = A[16,K] @ W[Ntot,K].T (+bias)(+add)
// W is packed bf16. Ntot must be a multiple of 64.
// ------------------------------------------------------------------
__device__ __forceinline__ void dec_gemm(const float* A, int K, const __bf16* W, long ldw,
                                         const float* bias, const float* add, long ldadd,
                                         float* out, long ldout, int Ntot) {
  int nwaves = blockDim.x >> 5, wave = uniform_(threadIdx.x >> 5);
  int lane = threadIdx.x & 31, half = lane >> 4, n = lane & 15;
  int gn = Ntot >> 6;
  v8f zero = {};
  for (int g = wave; g < gn; g += nwaves) {
    v8f acc[4] = {zero, zero, zero, zero};
    wmma_group4_bf(acc, A, K, 16, W, ldw, Ntot, g * 4, K);
#pragma unroll
    for (int j = 0; j < 4; ++j) {
      int n0 = (g * 4 + j) * 16;
      float bv = bias ? bias[n0 + n] : 0.0f;
#pragma unroll
      for (int v = 0; v < 8; ++v) {
        int m = v + half * 8;
        float x = acc[j][v] + bv;
        if (add) x += add[(long)m * ldadd + n0 + n];
        out[(long)m * ldout + n0 + n] = x;
      }
    }
  }
}

// ------------------------------------------------------------------
// persistent decoder scan (single workgroup, 32 steps): 2 GRU cells +
// coverage attention. Saves hid_t for the deferred vocab projection and
// cov_e_t for the pointer scatter; accumulates coverage loss.
// ------------------------------------------------------------------
__global__ __launch_bounds__(512) void k_dec_scan(
    const float* gx_emb,            // [T,B,3*H2]  (emb part of gx + dbi0)
    const __bf16* Wi0c,             // dWi0(bf16) + 300, ld = 1324 (ctx columns)
    const __bf16* Wh0, const float* bh0,
    const __bf16* Wi1, const float* bi1,
    const __bf16* Wh1, const float* bh1,
    const __bf16* pW1, const float* pb1,
    const float* memd,              // [B,TS,H2]
    const float* h0i, const float* h1i,
    float* h0, float* h1, float* ctx, float* cov,
    float* gA, float* gB,
    float* cov_e_all,               // [T,B,TS]
    float* hid_all,                 // [T*B, H2]
    float* covl_out) {
  const int Bn = Bn_, TS = TSn_, H2 = H2n_, G = 3 * H2n_, Tn = Tn_;
  __shared__ float sE[Bn_ * TSn_];        // energies -> attn (25.6 KB)
  __shared__ float sLossB[Bn_];
  __shared__ float sLoss;
  int tid = threadIdx.x;
  for (int i = tid; i < Bn * H2; i += blockDim.x) { h0[i] = h0i[i]; h1[i] = h1i[i]; ctx[i] = 0.0f; }
  for (int i = tid; i < Bn * TS; i += blockDim.x) cov[i] = 0.0f;
  if (tid == 0) sLoss = 0.0f;
  __syncthreads();

  for (int t = 0; t < Tn; ++t) {
    // cell 0: gx = gx_emb[t] + ctx @ Wi0[:,300:].T ; gh = h0 @ Wh0.T + bh0
    dec_gemm(ctx, H2, Wi0c, EMBn + H2n_, nullptr,
             gx_emb + (long)t * Bn * G, G, gA, G, G);
    dec_gemm(h0, H2, Wh0, H2, bh0, nullptr, 0, gB, G, G);
    __syncthreads();
    for (int i = tid; i < Bn * H2; i += blockDim.x) {
      int b = i / H2, j = i % H2;
      const float* ga = gA + (long)b * G;
      const float* gb = gB + (long)b * G;
      float r = sigmoidf_(ga[j] + gb[j]);
      float z = sigmoidf_(ga[H2 + j] + gb[H2 + j]);
      float n = tanhf(ga[2 * H2 + j] + r * gb[2 * H2 + j]);
      h0[i] = (1.0f - z) * n + z * h0[i];
    }
    __syncthreads();
    // cell 1
    dec_gemm(h0, H2, Wi1, H2, bi1, nullptr, 0, gA, G, G);
    dec_gemm(h1, H2, Wh1, H2, bh1, nullptr, 0, gB, G, G);
    __syncthreads();
    for (int i = tid; i < Bn * H2; i += blockDim.x) {
      int b = i / H2, j = i % H2;
      const float* ga = gA + (long)b * G;
      const float* gb = gB + (long)b * G;
      float r = sigmoidf_(ga[j] + gb[j]);
      float z = sigmoidf_(ga[H2 + j] + gb[H2 + j]);
      float n = tanhf(ga[2 * H2 + j] + r * gb[2 * H2 + j]);
      h1[i] = (1.0f - z) * n + z * h1[i];
    }
    __syncthreads();
    // energy + coverage-scaled energy (float4 dot products)
    for (int i = tid; i < Bn * TS; i += blockDim.x) {
      int b = i / TS, s = i % TS;
      const float4* hv = (const float4*)(h1 + (long)b * H2);
      const float4* mv = (const float4*)(memd + ((long)b * TS + s) * H2);
      float acc = 0.0f;
      for (int d = 0; d < H2 / 4; ++d) {
        float4 hh = hv[d], mm = mv[d];
        acc += hh.x * mm.x + hh.y * mm.y + hh.z * mm.z + hh.w * mm.w;
      }
      float ce = acc * (1.0f - cov[i]);
      sE[i] = ce;
      cov_e_all[(long)t * Bn * TS + i] = ce;
    }
    __syncthreads();
    // per-row softmax (wave w handles batch b = w), coverage loss + update
    {
      int wave = tid >> 5, lane = tid & 31;
      if (wave < Bn) {
        float* row = sE + wave * TS;
        float* cvr = cov + wave * TS;
        float mx = -1e30f;
        for (int s = lane; s < TS; s += 32) mx = fmaxf(mx, row[s]);
        for (int o = 16; o; o >>= 1) mx = fmaxf(mx, __shfl_xor(mx, o, 32));
        float sum = 0.0f;
        for (int s = lane; s < TS; s += 32) { float e = __expf(row[s] - mx); row[s] = e; sum += e; }
        for (int o = 16; o; o >>= 1) sum += __shfl_xor(sum, o, 32);
        float inv = 1.0f / sum;
        float part = 0.0f;
        for (int s = lane; s < TS; s += 32) {
          float a = row[s] * inv; row[s] = a;
          float cv = cvr[s];
          part += fminf(a, cv);
          cvr[s] = cv + a;
        }
        for (int o = 16; o; o >>= 1) part += __shfl_xor(part, o, 32);
        if (lane == 0) sLossB[wave] = part;
      }
    }
    __syncthreads();
    if (tid == 0) { float q = 0.0f; for (int b = 0; b < Bn; ++b) q += sLossB[b]; sLoss += q; }
    // new context (float4 weighted gather)
    for (int i4 = tid; i4 < Bn * (H2 / 4); i4 += blockDim.x) {
      int b = i4 / (H2 / 4), dq = i4 % (H2 / 4);
      const float* arow = sE + b * TS;
      const float4* mbase = (const float4*)(memd + (long)b * TS * H2) + dq;
      float4 acc = make_float4(0.f, 0.f, 0.f, 0.f);
      for (int s = 0; s < TS; ++s) {
        float w = arow[s];
        float4 m = mbase[(long)s * (H2 / 4)];
        acc.x += w * m.x; acc.y += w * m.y; acc.z += w * m.z; acc.w += w * m.w;
      }
      ((float4*)ctx)[i4] = acc;
    }
    __syncthreads();
    // hid = tanh([h1 | ctx] @ pW1.T + pb1)  (split-K concat, WMMA, bf16 W)
    {
      int nwaves = blockDim.x >> 5, wv = uniform_(tid >> 5);
      int lane = tid & 31, half = lane >> 4, n = lane & 15;
      v8f zero = {};
      int gn2 = H2 >> 6;
      for (int g = wv; g < gn2; g += nwaves) {
        v8f acc[4] = {zero, zero, zero, zero};
        wmma_group4_bf(acc, h1,  H2, 16, pW1,      2 * H2, H2, g * 4, H2);
        wmma_group4_bf(acc, ctx, H2, 16, pW1 + H2, 2 * H2, H2, g * 4, H2);
#pragma unroll
        for (int j = 0; j < 4; ++j) {
          int n0 = (g * 4 + j) * 16;
#pragma unroll
          for (int v = 0; v < 8; ++v) {
            int m = v + half * 8;
            float x = tanhf(acc[j][v] + pb1[n0 + n]);
            hid_all[((long)t * Bn + m) * H2 + n0 + n] = x;
          }
        }
      }
    }
    __syncthreads();
  }
  if (tid == 0) covl_out[0] = sLoss / (float)(Bn * Tn);
}

// ---------------- pointer-copy scatter / final combine ----------------
__global__ void k_fill(float* p, long n, float v) {
  long i = (long)blockIdx.x * blockDim.x + threadIdx.x;
  if (i < n) p[i] = v;
}

__global__ void k_scatter_max(const float* cov_e_t, const int* ext_ids, float* out_buf) {
  int i = blockIdx.x * blockDim.x + threadIdx.x;
  if (i >= Bn_ * TSn_) return;
  int b = i / TSn_;
  float v = cov_e_t[i];
  float* addr = out_buf + (long)b * EXTn_ + ext_ids[i];
  int old = __float_as_int(*addr);
  while (v > __int_as_float(old)) {
    int prev = atomicCAS((int*)addr, old, __float_as_int(v));
    if (prev == old) break;
    old = prev;
  }
}

__global__ void k_combine(const float* logits_t, const float* out_buf, float* lg, int t) {
  long i = (long)blockIdx.x * blockDim.x + threadIdx.x;
  const long tot = (long)Bn_ * EXTn_;
  if (i >= tot) return;
  int b = (int)(i / EXTn_), e = (int)(i % EXTn_);
  float ext = (e < Vn) ? logits_t[(long)b * Vn + e] : 0.0f;
  float o = out_buf[i];
  o = (o < -5.0e11f) ? 0.0f : o;           // out==-INF -> 0
  float r = ext + o;
  if (r == 0.0f) r = NEG_INF;              // lg==0 -> -INF
  lg[((long)b * Tn_ + t) * EXTn_ + e] = r;
}

// ------------------------------------------------------------------
extern "C" void kernel_launch(void* const* d_in, const int* in_sizes, int n_in,
                              void* d_out, int out_size, void* d_ws, size_t ws_size,
                              hipStream_t stream) {
  (void)in_sizes; (void)n_in; (void)out_size; (void)ws_size;
  const float* enc_emb  = (const float*)d_in[0];
  const float* ans_emb  = (const float*)d_in[1];
  const float* dec_emb  = (const float*)d_in[2];
  const float* eWi0     = (const float*)d_in[3];
  const float* eWh0     = (const float*)d_in[4];
  const float* ebi0     = (const float*)d_in[5];
  const float* ebh0     = (const float*)d_in[6];
  const float* eWi1     = (const float*)d_in[7];
  const float* eWh1     = (const float*)d_in[8];
  const float* ebi1     = (const float*)d_in[9];
  const float* ebh1     = (const float*)d_in[10];
  const float* trans_W  = (const float*)d_in[11];
  const float* trans_b  = (const float*)d_in[12];
  const float* upd_W    = (const float*)d_in[13];
  const float* gate_W   = (const float*)d_in[14];
  const float* dWi0     = (const float*)d_in[15];
  const float* dWh0     = (const float*)d_in[16];
  const float* dbi0     = (const float*)d_in[17];
  const float* dbh0     = (const float*)d_in[18];
  const float* dWi1     = (const float*)d_in[19];
  const float* dWh1     = (const float*)d_in[20];
  const float* dbi1     = (const float*)d_in[21];
  const float* dbh1     = (const float*)d_in[22];
  const float* dtrans_W = (const float*)d_in[23];
  const float* dtrans_b = (const float*)d_in[24];
  const float* pW1      = (const float*)d_in[25];
  const float* pb1      = (const float*)d_in[26];
  const float* pW2      = (const float*)d_in[27];
  const float* pb2      = (const float*)d_in[28];
  const int*   input_s  = (const int*)d_in[29];
  const int*   ext_ids  = (const int*)d_in[30];
  const int*   input_q  = (const int*)d_in[31];
  const int*   input_a  = (const int*)d_in[32];
  // d_in[33] = s_pos (unused by reference)

  // -------- workspace layout (floats, bump allocated) --------
  float* Wsp = (float*)d_ws;
  size_t off = 0;
  auto alloc = [&](size_t nf) { float* p = Wsp + off; off += nf; return p; };
  auto allocBf = [&](size_t ne) { __bf16* p = (__bf16*)(Wsp + off); off += (ne + 1) / 2; return p; };
  const long NB = (long)TSn_ * Bn_;                         // 6400
  float* X0    = alloc(NB * (EMBn + EEMBn));                // [6400,332]
  float* gxA   = alloc(NB * 3 * Hn_);                       // [6400,1536] (shared L0/L1)
  float* gxB   = alloc(NB * 3 * Hn_);
  float* y0    = alloc(NB * 2 * Hn_);                       // [TS,B,1024]
  float* y1    = alloc(NB * 2 * Hn_);
  float* obts  = alloc(NB * H2n_);                          // output -> self_out -> enc_out
  float* mem   = alloc(NB * H2n_);                          // memories -> mem_d
  float* energ = alloc((long)Bn_ * TSn_ * TSn_);            // energies -> scores
  float* ctx   = alloc(NB * H2n_);
  float* ghh   = alloc(2 * Bn_ * 3 * Hn_);
  float* hws   = alloc(2 * Bn_ * Hn_);                      // GRU hidden state (per dir)
  float* hl0   = alloc(Bn_ * H2n_);
  float* hl1   = alloc(Bn_ * H2n_);
  float* Xq    = alloc((long)Tn_ * Bn_ * EMBn);
  float* gxE   = alloc((long)Tn_ * Bn_ * 3 * H2n_);
  float* dh0   = alloc(Bn_ * H2n_);
  float* dh1   = alloc(Bn_ * H2n_);
  float* dctx  = alloc(Bn_ * H2n_);
  float* dcov  = alloc(Bn_ * TSn_);
  float* gA    = alloc(Bn_ * 3 * H2n_);
  float* gB    = alloc(Bn_ * 3 * H2n_);
  float* covE  = alloc((long)Tn_ * Bn_ * TSn_);
  float* hidA  = alloc((long)Tn_ * Bn_ * H2n_);
  float* logit = alloc((long)Tn_ * Bn_ * Vn);
  float* obuf  = alloc((long)Bn_ * EXTn_);
  // packed bf16 weights
  __bf16* eWi0b  = allocBf((size_t)2 * 3 * Hn_ * (EMBn + EEMBn));
  __bf16* eWh0b  = allocBf((size_t)2 * 3 * Hn_ * Hn_);
  __bf16* eWi1b  = allocBf((size_t)2 * 3 * Hn_ * 2 * Hn_);
  __bf16* eWh1b  = allocBf((size_t)2 * 3 * Hn_ * Hn_);
  __bf16* transb = allocBf((size_t)H2n_ * H2n_);
  __bf16* updb   = allocBf((size_t)H2n_ * 2 * H2n_);
  __bf16* gateb  = allocBf((size_t)H2n_ * 2 * H2n_);
  __bf16* dWi0b  = allocBf((size_t)3 * H2n_ * (EMBn + H2n_));
  __bf16* dWh0b  = allocBf((size_t)3 * H2n_ * H2n_);
  __bf16* dWi1b  = allocBf((size_t)3 * H2n_ * H2n_);
  __bf16* dWh1b  = allocBf((size_t)3 * H2n_ * H2n_);
  __bf16* dtransb= allocBf((size_t)H2n_ * H2n_);
  __bf16* pW1b   = allocBf((size_t)H2n_ * 2 * H2n_);
  __bf16* pW2b   = allocBf((size_t)Vn * H2n_);
  __bf16* membf  = allocBf((size_t)NB * H2n_);
  float* f_t  = y1;    // y1 dead after transpose
  float* g_t  = y0;    // y0 dead after layer-1 gx GEMMs
  float* eraw = gxA;   // gx buffers dead after layer-1 recurrence

  auto cvt = [&](const float* src, __bf16* dst, long n) {
    k_cvt_bf16<<<(unsigned)((n / 4 + 255) / 256 + 1), 256, 0, stream>>>(src, dst, n);
  };
  auto gemm = [&](const float* A, long lda, long sA, const void* Bm, long ldb, long sB,
                  int tB, int bf, const float* bias, float* C, long ldc, long sC,
                  int M, int N, int K, int act, int addC, int batch) {
    long groups = (long)((M + 15) / 16) * (((N + 15) / 16 + 3) / 4);
    dim3 grid((unsigned)((groups + 7) / 8), 1, (unsigned)batch);
    if (bf) k_gemm_t<1><<<grid, 256, 0, stream>>>(A, lda, sA, Bm, ldb, sB, tB, bias, C,
                                                  ldc, sC, M, N, K, act, addC);
    else    k_gemm_t<0><<<grid, 256, 0, stream>>>(A, lda, sA, Bm, ldb, sB, tB, bias, C,
                                                  ldc, sC, M, N, K, act, addC);
  };
  const int IN0 = EMBn + EEMBn;  // 332

  // ======== one-time weight packing to bf16 ========
  cvt(eWi0, eWi0b, (long)2 * 3 * Hn_ * IN0);
  cvt(eWh0, eWh0b, (long)2 * 3 * Hn_ * Hn_);
  cvt(eWi1, eWi1b, (long)2 * 3 * Hn_ * 2 * Hn_);
  cvt(eWh1, eWh1b, (long)2 * 3 * Hn_ * Hn_);
  cvt(trans_W, transb, (long)H2n_ * H2n_);
  cvt(upd_W, updb, (long)H2n_ * 2 * H2n_);
  cvt(gate_W, gateb, (long)H2n_ * 2 * H2n_);
  cvt(dWi0, dWi0b, (long)3 * H2n_ * (EMBn + H2n_));
  cvt(dWh0, dWh0b, (long)3 * H2n_ * H2n_);
  cvt(dWi1, dWi1b, (long)3 * H2n_ * H2n_);
  cvt(dWh1, dWh1b, (long)3 * H2n_ * H2n_);
  cvt(dtrans_W, dtransb, (long)H2n_ * H2n_);
  cvt(pW1, pW1b, (long)H2n_ * 2 * H2n_);
  cvt(pW2, pW2b, (long)Vn * H2n_);

  // ======== encoder ========
  { long tot = NB * IN0;
    k_embed_enc<<<(unsigned)((tot + 255) / 256), 256, 0, stream>>>(enc_emb, ans_emb, input_s, input_a, X0); }
  gemm(X0, IN0, 0, eWi0b, IN0, 0, 1, 1, ebi0, gxA, 3 * Hn_, 0, (int)NB, 3 * Hn_, IN0, 0, 0, 1);
  gemm(X0, IN0, 0, eWi0b + (long)3 * Hn_ * IN0, IN0, 0, 1, 1, ebi0 + 3 * Hn_, gxB, 3 * Hn_, 0,
       (int)NB, 3 * Hn_, IN0, 0, 0, 1);
  k_gru_layer<<<2, 512, 0, stream>>>(gxA, gxB, eWh0b, (long)3 * Hn_ * Hn_, ebh0, 3 * Hn_, y0, hl0, ghh, hws);
  gemm(y0, 2 * Hn_, 0, eWi1b, 2 * Hn_, 0, 1, 1, ebi1, gxA, 3 * Hn_, 0, (int)NB, 3 * Hn_, 2 * Hn_, 0, 0, 1);
  gemm(y0, 2 * Hn_, 0, eWi1b + (long)3 * Hn_ * 2 * Hn_, 2 * Hn_, 0, 1, 1, ebi1 + 3 * Hn_, gxB, 3 * Hn_, 0,
       (int)NB, 3 * Hn_, 2 * Hn_, 0, 0, 1);
  k_gru_layer<<<2, 512, 0, stream>>>(gxA, gxB, eWh1b, (long)3 * Hn_ * Hn_, ebh1, 3 * Hn_, y1, hl1, ghh, hws);
  { long tot = NB * H2n_;
    k_transpose<<<(unsigned)((tot + 255) / 256), 256, 0, stream>>>(y1, obts); }
  // memories (+ bf16 copy for the energies GEMM B-operand)
  gemm(obts, H2n_, 0, transb, H2n_, 0, 1, 1, trans_b, mem, H2n_, 0, (int)NB, H2n_, H2n_, 0, 0, 1);
  cvt(mem, membf, NB * H2n_);
  // gated self-attention
  gemm(obts, H2n_, (long)TSn_ * H2n_, membf, H2n_, (long)TSn_ * H2n_, 1, 1, nullptr,
       energ, TSn_, (long)TSn_ * TSn_, TSn_, TSn_, H2n_, 0, 0, Bn_);
  k_softmax_rows<<<(Bn_ * TSn_ + 7) / 8, 256, 0, stream>>>(energ, Bn_ * TSn_, TSn_);
  gemm(energ, TSn_, (long)TSn_ * TSn_, obts, H2n_, (long)TSn_ * H2n_, 0, 0, nullptr,
       ctx, H2n_, (long)TSn_ * H2n_, TSn_, H2n_, TSn_, 0, 0, Bn_);
  // f_t = tanh([out|ctx]@upd_W.T)  (split-K, second pass accumulates + activates)
  gemm(obts, H2n_, 0, updb, 2 * H2n_, 0, 1, 1, nullptr, f_t, H2n_, 0, (int)NB, H2n_, H2n_, 0, 0, 1);
  gemm(ctx, H2n_, 0, updb + H2n_, 2 * H2n_, 0, 1, 1, nullptr, f_t, H2n_, 0, (int)NB, H2n_, H2n_, 1, 1, 1);
  // g_t = sigmoid([out|ctx]@gate_W.T)
  gemm(obts, H2n_, 0, gateb, 2 * H2n_, 0, 1, 1, nullptr, g_t, H2n_, 0, (int)NB, H2n_, H2n_, 0, 0, 1);
  gemm(ctx, H2n_, 0, gateb + H2n_, 2 * H2n_, 0, 1, 1, nullptr, g_t, H2n_, 0, (int)NB, H2n_, H2n_, 2, 1, 1);
  { long tot = NB * H2n_;
    unsigned bl = (unsigned)((tot + 255) / 256);
    k_selfout<<<bl, 256, 0, stream>>>(obts, f_t, g_t);
    k_eraw<<<bl, 256, 0, stream>>>(mem, input_a, eraw);
    k_softmax_dim1<<<(Bn_ * H2n_ + 7) / 8, 256, 0, stream>>>(eraw, TSn_, H2n_);
    k_encout<<<bl, 256, 0, stream>>>(obts, eraw, mem); }
  // mem_d (overwrites memories buffer)
  gemm(obts, H2n_, 0, dtransb, H2n_, 0, 1, 1, dtrans_b, mem, H2n_, 0, (int)NB, H2n_, H2n_, 0, 0, 1);

  // ======== decoder ========
  { long tot = (long)Tn_ * Bn_ * EMBn;
    k_embed_dec<<<(unsigned)((tot + 255) / 256), 256, 0, stream>>>(dec_emb, input_q, Xq); }
  // emb part of cell-0 input projection (+ dbi0) for all 32 steps at once
  gemm(Xq, EMBn, 0, dWi0b, EMBn + H2n_, 0, 1, 1, dbi0, gxE, 3 * H2n_, 0,
       Tn_ * Bn_, 3 * H2n_, EMBn, 0, 0, 1);
  float* covl_out = (float*)d_out + (long)Bn_ * Tn_ * EXTn_;
  k_dec_scan<<<1, 512, 0, stream>>>(gxE, dWi0b + EMBn, dWh0b, dbh0, dWi1b, dbi1, dWh1b, dbh1,
                                    pW1b, pb1, mem, hl0, hl1,
                                    dh0, dh1, dctx, dcov, gA, gB, covE, hidA, covl_out);
  // deferred vocab projection: one [512,1024]x[1024,50000] GEMM,
  // pW2 in bf16 (100 MB, L2-resident) read once
  gemm(hidA, H2n_, 0, pW2b, H2n_, 0, 1, 1, pb2, logit, Vn, 0, Tn_ * Bn_, Vn, H2n_, 0, 0, 1);
  // pointer-copy merge per step
  for (int t = 0; t < Tn_; ++t) {
    long no = (long)Bn_ * EXTn_;
    unsigned bl = (unsigned)((no + 255) / 256);
    k_fill<<<bl, 256, 0, stream>>>(obuf, no, NEG_INF);
    k_scatter_max<<<(Bn_ * TSn_ + 255) / 256, 256, 0, stream>>>(covE + (long)t * Bn_ * TSn_, ext_ids, obuf);
    k_combine<<<bl, 256, 0, stream>>>(logit + (long)t * Bn_ * Vn, obuf, (float*)d_out, t);
  }
}
